// KWS_LSTM_bmm_25941602468229
// MI455X (gfx1250) — matrix operand order repaired and checked by reading, not compiled, not run on hardware
//
#include <hip/hip_runtime.h>
#include <hip/hip_bf16.h>

typedef __attribute__((ext_vector_type(16))) _Float16 v16h;
typedef __attribute__((ext_vector_type(8)))  float    v8f;

#define NB      8
#define SEQ     101
#define BATCH   256
#define IN_DIM  40
#define HID     200
#define G4      800     // 4*HID
#define KIH     64      // K of w_ih padded 40 -> 64  (2 chunks of 32)
#define KHH     224     // K of w_hh padded 200 -> 224 (7 chunks of 32)

// quant(w, 8, r=1.0): clip to +/-127/128, round-half-even to k/128
__device__ __forceinline__ float quantw(float w) {
    float x = fminf(fmaxf(w, -127.f / 128.f), 127.f / 128.f);
    return rintf(x * 128.f) * (1.f / 128.f);
}
// qp(v, a) = quant(pact(v,a), 8, a) ; pact = sign(v)*min(|v|,a)
__device__ __forceinline__ float qp8(float v, float a) {
    float p = copysignf(fminf(fabsf(v), a), v);
    float x = fminf(fmaxf(p / a, -127.f / 128.f), 127.f / 128.f);
    return rintf(x * 128.f) * (1.f / 128.f) * a;
}

// ---------------------------------------------------------------------------
// One-time weight fake-quant into workspace, transposed + K-padded so a B
// fragment (16 consecutive K of one output column) is one contiguous 32B load.
//   wihT[b][n][k] (k<KIH),  whhT[b][n][k] (k<KHH),  biasq[b][n] = q(b_ih)+q(b_hh)
// ---------------------------------------------------------------------------
__global__ void prequant_kernel(const float* __restrict__ w_ih,
                                const float* __restrict__ w_hh,
                                const float* __restrict__ b_ih,
                                const float* __restrict__ b_hh,
                                _Float16* __restrict__ wihT,
                                _Float16* __restrict__ whhT,
                                float* __restrict__ biasq) {
    long idx = (long)blockIdx.x * blockDim.x + threadIdx.x;
    const long nHH = (long)NB * G4 * KHH;
    const long nIH = (long)NB * G4 * KIH;
    if (idx < nHH) {
        int b = (int)(idx / (G4 * KHH));
        int rem = (int)(idx % (G4 * KHH));
        int n = rem / KHH, k = rem % KHH;
        whhT[idx] = (_Float16)((k < HID) ? quantw(w_hh[((long)b * HID + k) * G4 + n]) : 0.f);
    } else if (idx < nHH + nIH) {
        long j = idx - nHH;
        int b = (int)(j / (G4 * KIH));
        int rem = (int)(j % (G4 * KIH));
        int n = rem / KIH, k = rem % KIH;
        wihT[j] = (_Float16)((k < IN_DIM) ? quantw(w_ih[((long)b * IN_DIM + k) * G4 + n]) : 0.f);
    } else if (idx < nHH + nIH + (long)NB * G4) {
        long j = idx - nHH - nIH;
        biasq[j] = quantw(b_ih[j]) + quantw(b_hh[j]);
    }
}

// ---------------------------------------------------------------------------
// Recurrent kernel: grid (BATCH/16, NB). One workgroup owns a 16-row batch
// tile of one block; h (f16) and c (f32) live in LDS for all 101 steps.
// Gates[16,800] = xq@Wih + h@Whh + bias via v_wmma_f32_16x16x32_f16.
// ---------------------------------------------------------------------------
__global__ __launch_bounds__(256)
void lstm_kernel(const float* __restrict__ inp,
                 const _Float16* __restrict__ wihT,
                 const _Float16* __restrict__ whhT,
                 const float* __restrict__ biasq,
                 const float* __restrict__ a1,  const float* __restrict__ a3,
                 const float* __restrict__ a4,  const float* __restrict__ a5,
                 const float* __restrict__ a6,  const float* __restrict__ a7,
                 const float* __restrict__ a8,  const float* __restrict__ a9,
                 const float* __restrict__ a10, const float* __restrict__ a11,
                 const float* __restrict__ fw,  const float* __restrict__ fb,
                 const float* __restrict__ fa1, const float* __restrict__ fa2,
                 float* __restrict__ out) {
    __shared__ _Float16 hbuf[16][KHH];        // hidden state (padded K region = 0)
    __shared__ float    cbuf[16][HID];        // cell state
    __shared__ float    actb[4][16][HID];     // quantized gate activations i,j,f,o

    const int tid  = threadIdx.x;
    const int lane = tid & 31;                // wave32
    const int wave = tid >> 5;                // 8 waves
    const int b    = blockIdx.y;              // LSTM block
    const int r0   = blockIdx.x * 16;         // batch row tile

    const float a1v = a1[b], a3v = a3[b], a4v = a4[b], a5v = a5[b], a6v = a6[b];
    const float a7v = a7[b], a8v = a8[b], a9v = a9[b], a10v = a10[b], a11v = a11[b];

    for (int i = tid; i < 16 * KHH; i += 256) (&hbuf[0][0])[i] = (_Float16)0.f;
    for (int i = tid; i < 16 * HID; i += 256) (&cbuf[0][0])[i] = 0.f;
    __syncthreads();

    const int M = lane & 15;                  // row of A / column of B and C
    const int g = lane >> 4;                  // lane half-group
    const _Float16* whhB = whhT + (size_t)b * G4 * KHH;
    const _Float16* wihB = wihT + (size_t)b * G4 * KIH;
    const float*    bqB  = biasq + b * G4;

    for (int t = 0; t < SEQ; ++t) {
        // ---- A fragments (contiguous K layout: V v holds K = 16*g+2*v) ----
        v16h hfrag[7], xfrag[2];
        const float* xrow = inp + ((size_t)t * BATCH + (r0 + M)) * IN_DIM;
#pragma unroll
        for (int c = 0; c < 7; ++c)
#pragma unroll
            for (int v = 0; v < 8; ++v) {
                int K = 32 * c + (g << 4) + (v << 1);
                hfrag[c][2 * v]     = hbuf[M][K];
                hfrag[c][2 * v + 1] = hbuf[M][K + 1];
            }
#pragma unroll
        for (int c = 0; c < 2; ++c)
#pragma unroll
            for (int v = 0; v < 8; ++v) {
                int K = 32 * c + (g << 4) + (v << 1);
                xfrag[c][2 * v]     = (_Float16)((K     < IN_DIM) ? qp8(xrow[K],     a1v) : 0.f);
                xfrag[c][2 * v + 1] = (_Float16)((K + 1 < IN_DIM) ? qp8(xrow[K + 1], a1v) : 0.f);
            }

        // ---- 50 N-tiles of 16 columns, round-robin over the 8 waves ----
        for (int nt = wave; nt < G4 / 16; nt += 8) {
            const int n = nt * 16 + M;        // output column held by this lane
            v8f acc = {};
            const _Float16* bI = wihB + (size_t)n * KIH + (g << 4);
#pragma unroll
            for (int c = 0; c < 2; ++c) {
                v16h bf = *(const v16h*)(bI + 32 * c);
                acc = __builtin_amdgcn_wmma_f32_16x16x32_f16(false, xfrag[c], false, bf,
                                                             (short)0, acc, false, false);
            }
            const _Float16* bH = whhB + (size_t)n * KHH + (g << 4);
            if (nt + 8 < G4 / 16)             // gfx1250 global_prefetch for next tile
                __builtin_prefetch(bH + (size_t)128 * KHH, 0, 1);
#pragma unroll
            for (int c = 0; c < 7; ++c) {
                v16h bf = *(const v16h*)(bH + 32 * c);
                acc = __builtin_amdgcn_wmma_f32_16x16x32_f16(false, hfrag[c], false, bf,
                                                             (short)0, acc, false, false);
            }
            // Gate type is uniform per lane (column fixed across the 8 C VGPRs)
            const int   type = n / HID;                 // 0:i 1:j 2:f 3:o
            const int   hh   = n - type * HID;
            const float ag   = (type == 0) ? a4v : (type == 1) ? a5v
                             : (type == 2) ? a3v : a6v;
            const float bq   = bqB[n];
#pragma unroll
            for (int v = 0; v < 8; ++v) {               // C layout: M = v + 8*g
                float gv = acc[v] + bq;
                float av = (type == 1) ? tanhf(gv) : 1.f / (1.f + expf(-gv));
                actb[type][v + (g << 3)][hh] = qp8(av, ag);
            }
        }
        __syncthreads();

        // ---- pointwise cell/hidden update ----
        for (int i = tid; i < 16 * HID; i += 256) {
            int m = i / HID, hh = i - m * HID;
            float ig = actb[0][m][hh], jg = actb[1][m][hh];
            float fg = actb[2][m][hh], og = actb[3][m][hh];
            float cc = cbuf[m][hh];
            float gc = qp8(cc * fg, a7v);
            float ai = qp8(ig * jg, a8v);
            float nc = qp8(gc + ai, a9v);
            float ac = qp8(tanhf(nc), a10v);
            float nh = qp8(ac * og, a11v);
            cbuf[m][hh] = nc;
            hbuf[m][hh] = (_Float16)nh;                 // k*a11/128: exact in f16
        }
        __syncthreads();
    }

    // ---- final FC (200 -> 2) + 12-class stacking ----
    if (tid < 16) {
        const float fa1v = fa1[b], fa2v = fa2[b];
        float s0 = quantw(fb[b * 2 + 0]);
        float s1 = quantw(fb[b * 2 + 1]);
        for (int h = 0; h < HID; ++h) {
            float z = qp8((float)hbuf[tid][h], fa1v);
            s0 += z * quantw(fw[(b * HID + h) * 2 + 0]);
            s1 += z * quantw(fw[(b * HID + h) * 2 + 1]);
        }
        float o0 = qp8(s0, fa2v);
        float o1 = qp8(s1, fa2v);
        int r = r0 + tid;
        if (b < 4) {
            out[r * 12 + b] = o0 + o1;
        } else {
            out[r * 12 + 4 + 2 * (b - 4) + 0] = o0;
            out[r * 12 + 4 + 2 * (b - 4) + 1] = o1;
        }
    }
}

// ---------------------------------------------------------------------------
extern "C" void kernel_launch(void* const* d_in, const int* in_sizes, int n_in,
                              void* d_out, int out_size, void* d_ws, size_t ws_size,
                              hipStream_t stream) {
    (void)in_sizes; (void)n_in; (void)out_size; (void)ws_size;
    const float* inp  = (const float*)d_in[0];
    const float* w_ih = (const float*)d_in[1];
    const float* w_hh = (const float*)d_in[2];
    const float* b_ih = (const float*)d_in[3];
    const float* b_hh = (const float*)d_in[4];
    const float* a1  = (const float*)d_in[5];
    const float* a3  = (const float*)d_in[6];
    const float* a4  = (const float*)d_in[7];
    const float* a5  = (const float*)d_in[8];
    const float* a6  = (const float*)d_in[9];
    const float* a7  = (const float*)d_in[10];
    const float* a8  = (const float*)d_in[11];
    const float* a9  = (const float*)d_in[12];
    const float* a10 = (const float*)d_in[13];
    const float* a11 = (const float*)d_in[14];
    const float* fw  = (const float*)d_in[15];
    const float* fb  = (const float*)d_in[16];
    const float* fa1 = (const float*)d_in[17];
    const float* fa2 = (const float*)d_in[18];

    // workspace: wihT (f16) | whhT (f16) | biasq (f32)  ~3.7 MB total
    _Float16* wihT = (_Float16*)d_ws;
    _Float16* whhT = wihT + (size_t)NB * G4 * KIH;
    float*    biasq = (float*)(whhT + (size_t)NB * G4 * KHH);

    long total = (long)NB * G4 * KHH + (long)NB * G4 * KIH + (long)NB * G4;
    prequant_kernel<<<dim3((unsigned)((total + 255) / 256)), dim3(256), 0, stream>>>(
        w_ih, w_hh, b_ih, b_hh, wihT, whhT, biasq);

    dim3 grid(BATCH / 16, NB);   // 16 batch tiles x 8 blocks = 128 workgroups
    lstm_kernel<<<grid, dim3(256), 0, stream>>>(
        inp, wihT, whhT, biasq,
        a1, a3, a4, a5, a6, a7, a8, a9, a10, a11,
        fw, fb, fa1, fa2, (float*)d_out);
}
